// NTXentLoss_2611340116258
// MI455X (gfx1250) — compile-verified, matrix operand order checked by reading, not run on hardware
//
#include <hip/hip_runtime.h>
#include <hip/hip_bf16.h>

// ---------------------------------------------------------------------------
// NT-Xent loss, fused for MI455X (gfx1250, wave32, WMMA, async-to-LDS).
//   logits = (zj/||zj||) @ (zi/||zi||)^T / 0.1   (8192 x 8192, K=128)
//   loss_i = logsumexp([pos_i, logits[i, negs]]) - pos_i ; out = mean(loss)
// Logits are bounded: |10*cos| <= 10  ->  fixed-shift softmax (m = 10.0),
// exactly one raw v_exp_f32 per logit element (arg in [-21, 0.2], so the
// libm denormal fixup is provably dead -> use __builtin_amdgcn_exp2f).
// ---------------------------------------------------------------------------

typedef _Float16 v16h __attribute__((ext_vector_type(16)));
typedef _Float16 h8   __attribute__((ext_vector_type(8)));
typedef _Float16 h4   __attribute__((ext_vector_type(4)));
typedef float    v8f  __attribute__((ext_vector_type(8)));
typedef int      v4i  __attribute__((ext_vector_type(4)));

// async-to-LDS builtin parameter types (from hipcc diagnostic: param 0 is
// "int __vector(4) __device__ *" i.e. non-const int4 in global AS).
typedef v4i __attribute__((address_space(1))) gv4i;
typedef v4i __attribute__((address_space(3))) lv4i;

#define TEMP_INV   10.0f          // 1 / temperature(0.1)
#define SHIFT      10.0f          // fixed logsumexp shift: logits <= 10
#define SHIFT_L2E  14.4269504089f // SHIFT * log2(e)
#define CSPLIT     8              // column splits (grid.y)
#define DIM        128            // embedding dim

#if __has_builtin(__builtin_amdgcn_global_load_async_to_lds_b128)
#define HAVE_ASYNC_LDS 1
#else
#define HAVE_ASYNC_LDS 0
#endif

#if __has_builtin(__builtin_amdgcn_exp2f)
#define EXP2_RAW(x) __builtin_amdgcn_exp2f(x)
#else
#define EXP2_RAW(x) exp2f(x)
#endif

#if __has_builtin(__builtin_amdgcn_logf)
#define LOG2_RAW(x) __builtin_amdgcn_logf(x)   // raw v_log_f32: log2(x)
#define HAVE_LOG_RAW 1
#else
#define HAVE_LOG_RAW 0
#endif

// ---------------------------------------------------------------------------
// Kernel 1: L2-normalize rows of zis/zjs, emit f16 copies; zero the output.
// ---------------------------------------------------------------------------
__global__ __launch_bounds__(256)
void k_prep(const float* __restrict__ zis, const float* __restrict__ zjs,
            _Float16* __restrict__ zih, _Float16* __restrict__ zjh,
            float* __restrict__ out, int B)
{
    const int lane = threadIdx.x & 31;
    const int wave = threadIdx.x >> 5;
    const int row  = blockIdx.x * 8 + wave;
    if (blockIdx.x == 0 && threadIdx.x == 0) out[0] = 0.0f;
    if (row >= B) return;

    {
        float4 v = ((const float4*)zis)[row * 32 + lane];
        float ss = v.x * v.x + v.y * v.y + v.z * v.z + v.w * v.w;
        #pragma unroll
        for (int off = 16; off; off >>= 1) ss += __shfl_xor(ss, off, 32);
        float rn = rsqrtf(ss);
        h4 o; o[0] = (_Float16)(v.x * rn); o[1] = (_Float16)(v.y * rn);
              o[2] = (_Float16)(v.z * rn); o[3] = (_Float16)(v.w * rn);
        ((h4*)zih)[row * 32 + lane] = o;
    }
    {
        float4 v = ((const float4*)zjs)[row * 32 + lane];
        float ss = v.x * v.x + v.y * v.y + v.z * v.z + v.w * v.w;
        #pragma unroll
        for (int off = 16; off; off >>= 1) ss += __shfl_xor(ss, off, 32);
        float rn = rsqrtf(ss);
        h4 o; o[0] = (_Float16)(v.x * rn); o[1] = (_Float16)(v.y * rn);
              o[2] = (_Float16)(v.z * rn); o[3] = (_Float16)(v.w * rn);
        ((h4*)zjh)[row * 32 + lane] = o;
    }
}

// ---------------------------------------------------------------------------
// Kernel 2: fused WMMA GEMM tile + masked fixed-shift exp-sum.
// 256 thr = 8 waves; each wave owns a 16-row strip, A fragments resident in
// VGPRs for the whole kernel. zi tiles (64 cols x 128) double-buffered in LDS
// via GLOBAL_LOAD_ASYNC_TO_LDS_B128 (ASYNCcnt) when available.
// grid = (B/128 row blocks, CSPLIT column splits).
// ---------------------------------------------------------------------------
__global__ __launch_bounds__(256)
void k_logits_softmax(const _Float16* __restrict__ zih,
                      const _Float16* __restrict__ zjh,
                      const int* __restrict__ ilab, const int* __restrict__ jlab,
                      float* __restrict__ s_part, int B)
{
#if HAVE_ASYNC_LDS
    __shared__ __align__(16) _Float16 Bsh[2][64 * DIM];   // 2 x 16 KB
#else
    __shared__ __align__(16) _Float16 Bsh[1][64 * DIM];   // 16 KB
#endif

    const int tid  = threadIdx.x;
    const int lane = tid & 31;
    const int wave = tid >> 5;
    const int rowBase = blockIdx.x * 128;
    const int split   = blockIdx.y;
    const int colsPerSplit = B / CSPLIT;
    const int colBase = split * colsPerSplit;
    const int nT      = colsPerSplit / 64;

    // ---- per-wave A fragments straight from global (ISA 16-bit 16x32 layout)
    // lane 0-15: M=lane,   K = 0..7  and 16..23   (kb = 0)
    // lane16-31: M=lane-16,K = 8..15 and 24..31   (kb = 8)
    const int mrow = lane & 15;
    const int kb   = (lane >> 4) * 8;
    const int hi8  = (lane >> 4) * 8;            // C/D layout: row M = r + hi8
    const _Float16* arow = zjh + (size_t)(rowBase + wave * 16 + mrow) * DIM;
    v16h afrag[4];
    #pragma unroll
    for (int c = 0; c < 4; ++c) {
        h8 lo = *(const h8*)(arow + c * 32 + kb);
        h8 hi = *(const h8*)(arow + c * 32 + kb + 16);
        afrag[c] = __builtin_shufflevector(lo, hi,
            0, 1, 2, 3, 4, 5, 6, 7, 8, 9, 10, 11, 12, 13, 14, 15);
    }
    int jl[8];
    #pragma unroll
    for (int r = 0; r < 8; ++r) jl[r] = jlab[rowBase + wave * 16 + r + hi8];

    float s_run[8];
    #pragma unroll
    for (int r = 0; r < 8; ++r) s_run[r] = 0.f;

    // ---- tile copy slots: thread -> (col cn, 32 halfs at cks) --------------
    const int ccn = tid >> 2;
    const int cks = (tid & 3) * 32;

#if HAVE_ASYNC_LDS
    auto issue_tile = [&](int t, int buf) {
        const _Float16* g = zih + (size_t)(colBase + t * 64 + ccn) * DIM + cks;
        _Float16* l = &Bsh[buf][ccn * DIM + cks];
        #pragma unroll
        for (int q = 0; q < 4; ++q)
            __builtin_amdgcn_global_load_async_to_lds_b128(
                (gv4i*)(g + q * 8),
                (lv4i*)(l + q * 8),
                0, 0);
    };
    issue_tile(0, 0);
#else
    h8 pre[4];
    auto load_regs = [&](int t) {
        const _Float16* g = zih + (size_t)(colBase + t * 64 + ccn) * DIM + cks;
        #pragma unroll
        for (int q = 0; q < 4; ++q) pre[q] = *(const h8*)(g + q * 8);
    };
    load_regs(0);
#endif

    for (int t = 0; t < nT; ++t) {
#if HAVE_ASYNC_LDS
        const int cur = t & 1;
        if (t + 1 < nT) {
            issue_tile(t + 1, 1 - cur);
            asm volatile("s_wait_asynccnt 4" ::: "memory");  // tile t landed
        } else {
            asm volatile("s_wait_asynccnt 0" ::: "memory");
        }
        __syncthreads();                  // all waves' copies visible
#else
        const int cur = 0;
        __syncthreads();                  // previous tile fully consumed
        #pragma unroll
        for (int q = 0; q < 4; ++q)
            *(h8*)&Bsh[0][ccn * DIM + cks + q * 8] = pre[q];
        if (t + 1 < nT) load_regs(t + 1);
        __syncthreads();
#endif

        // prefetch the 4 column-label groups for this tile
        int iln4[4];
        #pragma unroll
        for (int sub = 0; sub < 4; ++sub)
            iln4[sub] = ilab[colBase + t * 64 + sub * 16 + (lane & 15)];

        #pragma unroll
        for (int sub = 0; sub < 4; ++sub) {
            const int cn = sub * 16 + (lane & 15);
            v8f acc = {0.f, 0.f, 0.f, 0.f, 0.f, 0.f, 0.f, 0.f};
            #pragma unroll
            for (int c = 0; c < 4; ++c) {
                h8 lo = *(const h8*)&Bsh[cur][cn * DIM + c * 32 + kb];
                h8 hi = *(const h8*)&Bsh[cur][cn * DIM + c * 32 + kb + 16];
                v16h bfrag = __builtin_shufflevector(lo, hi,
                    0, 1, 2, 3, 4, 5, 6, 7, 8, 9, 10, 11, 12, 13, 14, 15);
                acc = __builtin_amdgcn_wmma_f32_16x16x32_f16(
                    false, afrag[c], false, bfrag, (short)0, acc, false, false);
            }
            // fixed-shift: s += exp(logit-10) = exp2(acc*10*log2e - 10*log2e)
            // arg in [-21, 0.2] -> raw v_exp_f32, no denorm fixup needed
            const int iln = iln4[sub];
            #pragma unroll
            for (int r = 0; r < 8; ++r) {
                float e = EXP2_RAW(fmaf(acc[r], SHIFT_L2E, -SHIFT_L2E));
                s_run[r] += (iln == jl[r]) ? 0.f : e;
            }
        }
#if HAVE_ASYNC_LDS
        __syncthreads();                  // done reading Bsh[cur] before reuse
#endif
    }

    // ---- sum partials across the 16 lanes holding each row ----------------
    #pragma unroll
    for (int r = 0; r < 8; ++r) {
        float s = s_run[r];
        #pragma unroll
        for (int off = 1; off < 16; off <<= 1) s += __shfl_xor(s, off, 32);
        if ((lane & 15) == 0) {
            int row = rowBase + wave * 16 + r + hi8;
            s_part[row * CSPLIT + split] = s;
        }
    }
}

// ---------------------------------------------------------------------------
// Kernel 3: per-row finalize (one wave per row).
// loss = SHIFT + log(exp(pos-SHIFT) + sum_negs) - pos
// ---------------------------------------------------------------------------
__global__ __launch_bounds__(256)
void k_finalize(const _Float16* __restrict__ zih, const _Float16* __restrict__ zjh,
                const float* __restrict__ s_part,
                const float* __restrict__ weights, const int* __restrict__ d_idx,
                float* __restrict__ out, int B)
{
    const int lane = threadIdx.x & 31;
    const int wave = threadIdx.x >> 5;
    const int row  = blockIdx.x * 8 + wave;
    if (row >= B) return;
    const int idx  = d_idx[0];
    const int prow = idx + row;

    h4 aj = ((const h4*)zjh)[row * 32 + lane];
    h4 ai = ((const h4*)zih)[(size_t)prow * 32 + lane];
    float acc = (float)aj[0] * (float)ai[0] + (float)aj[1] * (float)ai[1]
              + (float)aj[2] * (float)ai[2] + (float)aj[3] * (float)ai[3];
    #pragma unroll
    for (int off = 16; off; off >>= 1) acc += __shfl_xor(acc, off, 32);
    float pos = acc * TEMP_INV;

    float s = 0.f;
    #pragma unroll
    for (int c = 0; c < CSPLIT; ++c) s += s_part[row * CSPLIT + c];

    // denom = exp(pos-SHIFT) + s; loss = SHIFT + ln(denom) - pos
    float denom = EXP2_RAW((pos - SHIFT) * 1.44269504089f) + s;
#if HAVE_LOG_RAW
    float lnd = LOG2_RAW(denom) * 0.69314718056f;   // log2 -> ln
#else
    float lnd = __logf(denom);
#endif
    float loss = SHIFT + lnd - pos;
    float w    = weights[prow];
    loss = (loss * w) / w;                      // faithful to reference
    if (lane == 0) atomicAdd(out, loss / (float)B);
}

// ---------------------------------------------------------------------------
extern "C" void kernel_launch(void* const* d_in, const int* in_sizes, int n_in,
                              void* d_out, int out_size, void* d_ws, size_t ws_size,
                              hipStream_t stream)
{
    const float* zis     = (const float*)d_in[0];
    const float* zjs     = (const float*)d_in[1];
    const int*   ilab    = (const int*)d_in[2];
    const int*   jlab    = (const int*)d_in[3];
    const float* weights = (const float*)d_in[4];
    const int*   idx     = (const int*)d_in[5];
    float*       out     = (float*)d_out;

    const int B = in_sizes[2];                 // 8192
    char* ws = (char*)d_ws;
    _Float16* zih    = (_Float16*)ws;                              // B*D*2 bytes
    _Float16* zjh    = (_Float16*)(ws + (size_t)B * DIM * 2);      // B*D*2 bytes
    float*    s_part = (float*)(ws + (size_t)B * DIM * 4);         // B*CSPLIT*4

    k_prep<<<B / 8, 256, 0, stream>>>(zis, zjs, zih, zjh, out, B);

    dim3 grid(B / 128, CSPLIT);
    k_logits_softmax<<<grid, 256, 0, stream>>>(zih, zjh, ilab, jlab, s_part, B);

    k_finalize<<<B / 8, 256, 0, stream>>>(zih, zjh, s_part, weights, idx, out, B);
}
